// Attention_56564719288556
// MI455X (gfx1250) — matrix operand order repaired and checked by reading, not compiled, not run on hardware
//
#include <hip/hip_runtime.h>

typedef __attribute__((ext_vector_type(16))) __bf16 v16bf;
typedef __attribute__((ext_vector_type(8)))  float  v8f;

#define NH    32
#define NKVH  8
#define HD    128
#define GROUP 4
#define B_    8
#define LQ_   128
#define CTX   2048
#define PS    16
#define KT    32     // tokens per key tile = 2 pages = 1 causal block
#define KSTRB 136    // Khi/Klo row stride in bf16 (272B, 16B-aligned, bank-spread)
#define VSTRB 40     // VThi/VTlo row stride in bf16 (80B, 16B-aligned, bank-spread)
#define PSTR  36     // Pbuf row stride in f32 (bank-spread, 16B-aligned spans)

static __device__ __forceinline__ v8f wmma_bf16(v16bf a, v16bf b, v8f c) {
    return __builtin_amdgcn_wmma_f32_16x16x32_bf16(
        false, a, false, b, (short)0, c, false, false);
}

__global__ __launch_bounds__(256)
void pattn_prefill_bf16x3(const float* __restrict__ q,
                          const float* __restrict__ knew,
                          const float* __restrict__ vnew,
                          const float* __restrict__ kcache,
                          const float* __restrict__ vcache,
                          const int*   __restrict__ ptab,
                          float* __restrict__ out)
{
    __shared__ __bf16 Khi[KT][KSTRB];     // [token][d]   hi part
    __shared__ __bf16 Klo[KT][KSTRB];     // [token][d]   lo part
    __shared__ __bf16 VThi[HD][VSTRB];    // [d][token]   hi part (transposed)
    __shared__ __bf16 VTlo[HD][VSTRB];    // [d][token]   lo part
    __shared__ float  Pbuf[8][16][PSTR];  // per-wave P relayout scratch [wave][M][key]

    const int tid  = threadIdx.x;
    const int lane = tid & 31;
    const int w    = tid >> 5;       // wave 0..7
    const int hg   = w & 3;          // head within GQA group
    const int qsub = w >> 2;         // 16-query half of the 32-query block

    const int wg  = blockIdx.x;      // 256 = B * NKVH * 4
    const int qb  = wg & 3;
    const int kvh = (wg >> 2) & 7;
    const int b   = wg >> 5;
    const int head = kvh * GROUP + hg;

    const int n    = lane & 15;
    const int hsel = lane >> 4;

    // ---- Q fragments: pre-scaled, bf16 hi/lo split, WMMA 16-bit A-layout ----
    // element e of v16bf -> K = 16*hsel + e
    const float scale = 0.08838834764831845f;  // 1/sqrt(128)
    const int qrow = qb * 32 + qsub * 16 + n;
    const float* qptr = q + (size_t)(b * LQ_ + qrow) * (NH * HD) + head * HD;
    v16bf qhi[4], qlo[4];
#pragma unroll
    for (int kk = 0; kk < 4; ++kk) {
        v16bf h, l;
#pragma unroll
        for (int e = 0; e < 16; ++e) {
            const int d = 32 * kk + 16 * hsel + e;
            const float x  = qptr[d] * scale;
            const __bf16 xh = (__bf16)x;
            h[e] = xh;
            l[e] = (__bf16)(x - (float)xh);
        }
        qhi[kk] = h; qlo[kk] = l;
    }

    // ---- flash accumulators ----
    v8f O[8];
#pragma unroll
    for (int c = 0; c < 8; ++c) O[c] = {};
    float m_i[8], l_i[8];
#pragma unroll
    for (int r = 0; r < 8; ++r) { m_i[r] = -1e30f; l_i[r] = 0.0f; }

    // block-causal mask is tile-granular: 64 cache tiles + (qb+1) new tiles
    const int ntiles = 65 + qb;

    // staging split: K row-major (coalesced), V column-major (transpose-friendly)
    const int krow = tid >> 3;          // 0..31 token
    const int kcol = (tid & 7) * 16;    // 16-float chunk of d
    const int vd   = tid >> 1;          // 0..127 d row
    const int vth  = tid & 1;           // which 16-token half (== page within tile)

    for (int t = 0; t < ntiles; ++t) {
        // ---- source pointers (one page == 16 tokens, tiles are page-aligned) ----
        const float* ksrc;
        if (t < 64) {
            const int page = ptab[b * 128 + 2 * t + (krow >> 4)];
            ksrc = kcache + ((size_t)page * PS + (krow & 15)) * (NKVH * HD) + (size_t)kvh * HD;
        } else {
            ksrc = knew + (size_t)(b * LQ_ + (t - 64) * KT + krow) * (NKVH * HD) + (size_t)kvh * HD;
        }
        const float* vsrc;  // token (vth*16 + i) at vsrc + i*NKVH*HD
        if (t < 64) {
            const int pageV = ptab[b * 128 + 2 * t + vth];
            vsrc = vcache + (size_t)pageV * PS * (NKVH * HD) + (size_t)kvh * HD + vd;
        } else {
            vsrc = vnew + (size_t)(b * LQ_ + (t - 64) * KT + vth * 16) * (NKVH * HD) + (size_t)kvh * HD + vd;
        }

        __syncthreads();   // previous tile fully consumed

        // ---- K staging: load 16 f32, split into bf16 hi/lo, contiguous store ----
        float kf[16];
#pragma unroll
        for (int i = 0; i < 4; ++i) {
            const float4 f = *(const float4*)(ksrc + kcol + 4 * i);
            kf[4 * i + 0] = f.x; kf[4 * i + 1] = f.y;
            kf[4 * i + 2] = f.z; kf[4 * i + 3] = f.w;
        }
        v16bf khv, klv;
#pragma unroll
        for (int i = 0; i < 16; ++i) {
            const __bf16 xh = (__bf16)kf[i];
            khv[i] = xh;
            klv[i] = (__bf16)(kf[i] - (float)xh);
        }
        *(v16bf*)(&Khi[krow][kcol]) = khv;
        *(v16bf*)(&Klo[krow][kcol]) = klv;

        // ---- V staging: gather one d across 16 tokens, store transposed ----
        float vf[16];
#pragma unroll
        for (int i = 0; i < 16; ++i) vf[i] = vsrc[(size_t)i * (NKVH * HD)];
        v16bf vhv, vlv;
#pragma unroll
        for (int i = 0; i < 16; ++i) {
            const __bf16 xh = (__bf16)vf[i];
            vhv[i] = xh;
            vlv[i] = (__bf16)(vf[i] - (float)xh);
        }
        *(v16bf*)(&VThi[vd][vth * 16]) = vhv;
        *(v16bf*)(&VTlo[vd][vth * 16]) = vlv;

        // ---- prefetch next tile's K rows ----
        if (t + 1 < ntiles) {
            const float* pk;
            if (t + 1 < 64) {
                const int page2 = ptab[b * 128 + 2 * (t + 1) + (krow >> 4)];
                pk = kcache + ((size_t)page2 * PS + (krow & 15)) * (NKVH * HD) + (size_t)kvh * HD;
            } else {
                pk = knew + (size_t)(b * LQ_ + (t + 1 - 64) * KT + krow) * (NKVH * HD) + (size_t)kvh * HD;
            }
            __builtin_prefetch(pk + kcol, 0, 0);
        }

        __syncthreads();   // tile visible to all waves

        // ---- S = Q K^T : 2 key sub-tiles x 4 K-steps x 3 split terms ----
        v8f S0 = {}, S1 = {};
#pragma unroll
        for (int kk = 0; kk < 4; ++kk) {
            const int doff = 32 * kk + 16 * hsel;   // lane's contiguous d span
            v16bf kh_a = *(const v16bf*)(&Khi[n][doff]);
            v16bf kl_a = *(const v16bf*)(&Klo[n][doff]);
            v16bf kh_b = *(const v16bf*)(&Khi[16 + n][doff]);
            v16bf kl_b = *(const v16bf*)(&Klo[16 + n][doff]);
            S0 = wmma_bf16(qhi[kk], kh_a, S0);
            S0 = wmma_bf16(qhi[kk], kl_a, S0);
            S0 = wmma_bf16(qlo[kk], kh_a, S0);
            S1 = wmma_bf16(qhi[kk], kh_b, S1);
            S1 = wmma_bf16(qhi[kk], kl_b, S1);
            S1 = wmma_bf16(qlo[kk], kh_b, S1);
        }

        // ---- online softmax over 32 keys (rows: VGPR r -> M = r + 8*hsel) ----
#pragma unroll
        for (int r = 0; r < 8; ++r) {
            const float s0 = S0[r], s1 = S1[r];
            float tm = fmaxf(s0, s1);
            tm = fmaxf(tm, __shfl_xor(tm, 1, 16));
            tm = fmaxf(tm, __shfl_xor(tm, 2, 16));
            tm = fmaxf(tm, __shfl_xor(tm, 4, 16));
            tm = fmaxf(tm, __shfl_xor(tm, 8, 16));
            const float mnew  = fmaxf(m_i[r], tm);
            const float alpha = __expf(m_i[r] - mnew);
            const float p0 = __expf(s0 - mnew);
            const float p1 = __expf(s1 - mnew);
            float ps = p0 + p1;
            ps += __shfl_xor(ps, 1, 16);
            ps += __shfl_xor(ps, 2, 16);
            ps += __shfl_xor(ps, 4, 16);
            ps += __shfl_xor(ps, 8, 16);
            l_i[r] = l_i[r] * alpha + ps;
            m_i[r] = mnew;
            Pbuf[w][r + 8 * hsel][n]      = p0;
            Pbuf[w][r + 8 * hsel][16 + n] = p1;
#pragma unroll
            for (int c = 0; c < 8; ++c) O[c][r] *= alpha;
        }

        // ---- P: D-layout -> A-layout (LDS transpose) + bf16 hi/lo split ----
        // (same-wave DS RAW: DS ops are wave-ordered; compiler emits s_wait_dscnt)
        v16bf phi, plo;
        {
            const float* prow = &Pbuf[w][n][0];
            float pf[16];
            const float4 a0 = *(const float4*)(prow + 16 * hsel);
            const float4 a1 = *(const float4*)(prow + 16 * hsel + 4);
            const float4 a2 = *(const float4*)(prow + 16 * hsel + 8);
            const float4 a3 = *(const float4*)(prow + 16 * hsel + 12);
            pf[0] = a0.x; pf[1] = a0.y; pf[2]  = a0.z; pf[3]  = a0.w;
            pf[4] = a1.x; pf[5] = a1.y; pf[6]  = a1.z; pf[7]  = a1.w;
            pf[8] = a2.x; pf[9] = a2.y; pf[10] = a2.z; pf[11] = a2.w;
            pf[12] = a3.x; pf[13] = a3.y; pf[14] = a3.z; pf[15] = a3.w;
#pragma unroll
            for (int e = 0; e < 16; ++e) {
                const __bf16 xh = (__bf16)pf[e];
                phi[e] = xh;
                plo[e] = (__bf16)(pf[e] - (float)xh);
            }
        }

        // ---- O += P V : 8 d-chunks x 1 K-step (K=32) x 3 split terms ----
#pragma unroll
        for (int c = 0; c < 8; ++c) {
            v16bf vh = *(const v16bf*)(&VThi[16 * c + n][16 * hsel]);
            v16bf vl = *(const v16bf*)(&VTlo[16 * c + n][16 * hsel]);
            v8f acc = O[c];
            acc = wmma_bf16(phi, vh, acc);
            acc = wmma_bf16(phi, vl, acc);
            acc = wmma_bf16(plo, vh, acc);
            O[c] = acc;
        }
    }

    // ---- epilogue: divide by row sums and store ----
    const int orowbase = b * LQ_ + qb * 32 + qsub * 16;
#pragma unroll
    for (int r = 0; r < 8; ++r) {
        const int M = r + 8 * hsel;
        const float inv = 1.0f / l_i[r];
        float* orow = out + (size_t)(orowbase + M) * (NH * HD) + head * HD + n;
#pragma unroll
        for (int c = 0; c < 8; ++c)
            orow[16 * c] = O[c][r] * inv;
    }
}

extern "C" void kernel_launch(void* const* d_in, const int* in_sizes, int n_in,
                              void* d_out, int out_size, void* d_ws, size_t ws_size,
                              hipStream_t stream) {
    const float* q  = (const float*)d_in[0];
    const float* k  = (const float*)d_in[1];
    const float* v  = (const float*)d_in[2];
    const float* kc = (const float*)d_in[3];
    const float* vc = (const float*)d_in[4];
    const int*   pt = (const int*)d_in[5];
    // d_in[6] = ctx_len (2048), d_in[7] = block_size (32): fixed by the reference.
    float* out = (float*)d_out;

    dim3 grid(B_ * NKVH * (LQ_ / 32));   // 8 * 8 * 4 = 256 workgroups
    dim3 block(256);                     // 8 waves: 4 GQA heads x 2 query sub-tiles
    pattn_prefill_bf16x3<<<grid, block, 0, stream>>>(q, k, v, kc, vc, pt, out);
}
